// DeformableConv2d_24223615550264
// MI455X (gfx1250) — matrix operand order repaired and checked by reading, not compiled, not run on hardware
//
#include <hip/hip_runtime.h>

typedef __attribute__((ext_vector_type(16))) _Float16 v16h;
typedef __attribute__((ext_vector_type(8)))  _Float16 v8h;
typedef __attribute__((ext_vector_type(8)))  float    v8f;

#define BN 4
#define CI 64
#define CO 64
#define HH 128
#define WW 128
#define HW (HH*WW)          // 16384 = 1<<14
#define K2 9
#define KD 576              // C*K2, k index = tap*64 + c
#define TP 64               // pixels per block
#define NTHREADS 256

#define A_BYTES   (TP*KD*2)           // 73728
#define W_BYTES   (CO*KD*2)           // 73728
#define OM_BYTES  (TP*28*4)           // 7168 (27 used, padded)
#define SMEM_BYTES (A_BYTES + W_BYTES + OM_BYTES)   // 154624

#define CAT16(lo,hi) __builtin_shufflevector((lo),(hi),0,1,2,3,4,5,6,7,8,9,10,11,12,13,14,15)

__global__ __launch_bounds__(NTHREADS, 1)
void dcn_fused_kernel(const float* __restrict__ x,
                      const float* __restrict__ offset_w,
                      const float* __restrict__ offset_b,
                      const float* __restrict__ mask_w,
                      const float* __restrict__ mask_b,
                      const float* __restrict__ weight,
                      float* __restrict__ out)
{
    extern __shared__ char smem[];
    _Float16* Alds = (_Float16*)smem;                       // [TP][KD] f16
    _Float16* Wsh  = (_Float16*)(smem + A_BYTES);           // [<=CO][KD] f16 (reused)
    float*    om   = (float*)(smem + A_BYTES + W_BYTES);    // [TP][28] f32

    const int tid  = threadIdx.x;
    const int wv   = tid >> 5;       // wave 0..7 (wave32)
    const int lane = tid & 31;
    const int p0   = blockIdx.x * TP;

    // ---------- stage offset/mask conv weights: N=32 (18 offset + 9 mask + 5 pad) ----------
    for (int i = tid; i < 32*KD; i += NTHREADS) {
        int n = i / KD, k = i - n*KD;
        int tap = k >> 6, c = k & 63;
        float v = 0.f;
        if (n < 18)       v = offset_w[(n*CI + c)*K2 + tap];
        else if (n < 27)  v = mask_w[((n-18)*CI + c)*K2 + tap];
        Wsh[i] = (_Float16)v;
    }

    // ---------- build im2col A tile (f16), wave-uniform (pixel,tap), lane = channel ----------
    const int c   = tid & 63;        // channel owned by this lane
    const int grp = tid >> 6;        // 0..3 (wave-pair group)
    for (int pt = grp; pt < TP*K2; pt += 4) {
        int m = pt / K2, tap = pt - m*K2;
        int p = p0 + m;
        int b = p >> 14, hw = p & (HW-1);
        int h = hw >> 7, w = hw & (WW-1);
        int ti = tap / 3, tj = tap - ti*3;
        int y = h - 1 + ti, xx = w - 1 + tj;
        float v = 0.f;
        if ((unsigned)y < HH && (unsigned)xx < WW)
            v = x[((size_t)(b*CI + c) << 14) + (y << 7) + xx];
        Alds[m*KD + tap*64 + c] = (_Float16)v;
    }
    __syncthreads();

    // ---------- GEMM1: offsets + mask (M=64, N=32, K=576) via WMMA ----------
    {
        const int m_tile = wv >> 1;                 // 0..3
        const int n_tile = wv & 1;                  // 0..1
        v8f acc = {};
        const int arow  = (m_tile*16 + (lane & 15)) * KD;
        const int akoff = (lane >> 4) ? 8 : 0;
        const int brow  = (n_tile*16 + (lane & 15)) * KD;
        const int bkoff = (lane >> 4) ? 8 : 0;
        for (int kk = 0; kk < KD; kk += 32) {
            const _Float16* ap = &Alds[arow + kk + akoff];
            v8h alo = *(const v8h*)ap;
            v8h ahi = *(const v8h*)(ap + 16);
            v16h a  = CAT16(alo, ahi);
            const _Float16* bp = &Wsh[brow + kk + bkoff];
            v8h blo = *(const v8h*)bp;
            v8h bhi = *(const v8h*)(bp + 16);
            v16h bb = CAT16(blo, bhi);
            acc = __builtin_amdgcn_wmma_f32_16x16x32_f16(false, a, false, bb,
                                                         (short)0, acc, false, false);
        }
        int n = n_tile*16 + (lane & 15);
        if (n < 27) {
            int mloc = m_tile*16 + ((lane >> 4) << 3);
            for (int v = 0; v < 8; ++v) {
                float t = acc[v];
                if (n < 18) t += offset_b[n];
                else        t = 1.f / (1.f + __expf(-(t + mask_b[n-18])));
                om[(mloc + v)*28 + n] = t;
            }
        }
    }
    __syncthreads();

    // ---------- reload Wsh with main conv weights [O=64][KD] ----------
    for (int i = tid; i < CO*KD; i += NTHREADS) {
        int n = i / KD, k = i - n*KD;
        int tap = k >> 6, cc = k & 63;
        Wsh[i] = (_Float16)weight[(n*CI + cc)*K2 + tap];
    }

    // ---------- rebuild A as mask-modulated bilinear samples ----------
    for (int pt = grp; pt < TP*K2; pt += 4) {
        int m = pt / K2, tap = pt - m*K2;
        int p = p0 + m;
        int b = p >> 14, hw = p & (HW-1);
        int h = hw >> 7, w = hw & (WW-1);
        int ti = tap / 3, tj = tap - ti*3;
        float oy  = om[m*28 + 2*tap];
        float ox  = om[m*28 + 2*tap + 1];
        float msk = om[m*28 + 18 + tap];
        float py = oy + (float)(h - 1 + ti);
        float px = ox + (float)(w - 1 + tj);
        float y0f = floorf(py), x0f = floorf(px);
        float dy = py - y0f,   dx = px - x0f;
        int y0 = (int)y0f, x0 = (int)x0f;
        int y1 = y0 + 1,   x1 = x0 + 1;
        float w00 = (1.f-dy)*(1.f-dx), w01 = (1.f-dy)*dx;
        float w10 = dy*(1.f-dx),       w11 = dy*dx;
        w00 = ((unsigned)y0 < HH && (unsigned)x0 < WW) ? w00*msk : 0.f;
        w01 = ((unsigned)y0 < HH && (unsigned)x1 < WW) ? w01*msk : 0.f;
        w10 = ((unsigned)y1 < HH && (unsigned)x0 < WW) ? w10*msk : 0.f;
        w11 = ((unsigned)y1 < HH && (unsigned)x1 < WW) ? w11*msk : 0.f;
        int y0c = min(max(y0,0),HH-1), y1c = min(max(y1,0),HH-1);
        int x0c = min(max(x0,0),WW-1), x1c = min(max(x1,0),WW-1);
        const float* xb = x + ((size_t)(b*CI + c) << 14);
        float g = w00*xb[(y0c<<7)+x0c] + w01*xb[(y0c<<7)+x1c]
                + w10*xb[(y1c<<7)+x0c] + w11*xb[(y1c<<7)+x1c];
        Alds[m*KD + tap*64 + c] = (_Float16)g;
    }
    __syncthreads();

    // ---------- GEMM2: out (M=64 px, N=64 ch, K=576) via WMMA ----------
    {
        const int m_tile = wv >> 1;                 // 0..3
        const int n_half = wv & 1;                  // 0..1 -> N-tiles {2*n_half, 2*n_half+1}
        v8f acc0 = {}, acc1 = {};
        const int arow  = (m_tile*16 + (lane & 15)) * KD;
        const int akoff = (lane >> 4) ? 8 : 0;
        const int brow0 = ((n_half*2 + 0)*16 + (lane & 15)) * KD;
        const int brow1 = ((n_half*2 + 1)*16 + (lane & 15)) * KD;
        const int bkoff = (lane >> 4) ? 8 : 0;
        for (int kk = 0; kk < KD; kk += 32) {
            const _Float16* ap = &Alds[arow + kk + akoff];
            v8h alo = *(const v8h*)ap;
            v8h ahi = *(const v8h*)(ap + 16);
            v16h a  = CAT16(alo, ahi);
            const _Float16* b0p = &Wsh[brow0 + kk + bkoff];
            v16h b0 = CAT16(*(const v8h*)b0p, *(const v8h*)(b0p + 16));
            const _Float16* b1p = &Wsh[brow1 + kk + bkoff];
            v16h b1 = CAT16(*(const v8h*)b1p, *(const v8h*)(b1p + 16));
            acc0 = __builtin_amdgcn_wmma_f32_16x16x32_f16(false, a, false, b0,
                                                          (short)0, acc0, false, false);
            acc1 = __builtin_amdgcn_wmma_f32_16x16x32_f16(false, a, false, b1,
                                                          (short)0, acc1, false, false);
        }
        const int o0 = (n_half*2 + 0)*16 + (lane & 15);
        const int o1 = o0 + 16;
        const int mloc = m_tile*16 + ((lane >> 4) << 3);
        for (int v = 0; v < 8; ++v) {
            int p = p0 + mloc + v;
            int b = p >> 14, hw = p & (HW-1);
            out[((size_t)(b*CO + o0) << 14) + hw] = acc0[v];
            out[((size_t)(b*CO + o1) << 14) + hw] = acc1[v];
        }
    }
}

extern "C" void kernel_launch(void* const* d_in, const int* in_sizes, int n_in,
                              void* d_out, int out_size, void* d_ws, size_t ws_size,
                              hipStream_t stream) {
    (void)in_sizes; (void)n_in; (void)d_ws; (void)ws_size; (void)out_size;
    const float* x        = (const float*)d_in[0];
    const float* offset_w = (const float*)d_in[1];
    const float* offset_b = (const float*)d_in[2];
    const float* mask_w   = (const float*)d_in[3];
    const float* mask_b   = (const float*)d_in[4];
    const float* weight   = (const float*)d_in[5];
    float* out = (float*)d_out;

    hipFuncSetAttribute((const void*)dcn_fused_kernel,
                        hipFuncAttributeMaxDynamicSharedMemorySize, SMEM_BYTES);

    const int nblocks = (BN*HW) / TP;   // 1024
    dcn_fused_kernel<<<nblocks, NTHREADS, SMEM_BYTES, stream>>>(
        x, offset_w, offset_b, mask_w, mask_b, weight, out);
}
